// SparseAttention_18485539242942
// MI455X (gfx1250) — matrix operand order repaired and checked by reading, not compiled, not run on hardware
//
#include <hip/hip_runtime.h>
#include <hip/hip_bf16.h>

typedef __bf16 bf16;
typedef __attribute__((ext_vector_type(8)))  __bf16 v8bf;
typedef __attribute__((ext_vector_type(16))) __bf16 v16bf;
typedef __attribute__((ext_vector_type(8)))  float  v8f;

#define BQ   4
#define SEQ  4096
#define DMODEL 1024
#define NH   16
#define NKV  4
#define DH   64
#define CHUNK 256
#define HALFW 128
#define NCH  16

__device__ __forceinline__ v8f wmma_bf16(v16bf a, v16bf b, v8f c) {
  return __builtin_amdgcn_wmma_f32_16x16x32_bf16(false, a, false, b, (short)0, c, false, false);
}

// ---------------- conversion kernels ----------------
__global__ void f2bf_kernel(const float* __restrict__ in, bf16* __restrict__ out, size_t n) {
  for (size_t i = (size_t)blockIdx.x * blockDim.x + threadIdx.x; i < n;
       i += (size_t)gridDim.x * blockDim.x)
    out[i] = (bf16)in[i];
}

// in: row-major K x N fp32 ; out: row-major N x K bf16 (transposed)
__global__ void transpose_f2bf_kernel(const float* __restrict__ in, bf16* __restrict__ out,
                                      int K, int N) {
  size_t i = (size_t)blockIdx.x * blockDim.x + threadIdx.x;
  size_t total = (size_t)K * N;
  if (i >= total) return;
  int n = (int)(i % N);
  int k = (int)(i / N);
  out[(size_t)n * K + k] = (bf16)in[i];
}

// ---------------- fused GEMM (M=16384, K=1024, tile 128x64, async double-buffered) ----------------
// MODE 0: Q  -> bias, rmsnorm(qn), rope, bf16 [B,16,S,64]
// MODE 1: K  -> bias, rmsnorm(kn), rope, bf16 [B,4,S,64]
// MODE 2: V  -> bias, bf16 transposed [B,4,64,S]
// MODE 3: O  -> bias, fp32 row-major [16384,1024] (d_out)
template <int MODE>
__global__ __launch_bounds__(256, 1)
void gemm128_kernel(const bf16* __restrict__ A, const bf16* __restrict__ Bt,
                    const float* __restrict__ bias, const float* __restrict__ nw,
                    const float* __restrict__ cosT, const float* __restrict__ sinT,
                    bf16* __restrict__ obf, float* __restrict__ of, int N) {
  // double-buffered staging (row stride 80 elems = 160 B -> 32B-aligned rows)
  __shared__ __align__(32) bf16 sA[2][128 * 80];
  __shared__ __align__(32) bf16 sB[2][64 * 80];
  __shared__ float sC[(MODE < 2) ? 128 * 68 : 4];  // epilogue tile (modes 0/1)

  const int t    = threadIdx.x;
  const int wave = t >> 5;
  const int lane = t & 31;
  const int hf   = lane >> 4;
  const int l15  = lane & 15;
  const int mq   = wave & 3;     // 32-row group within 128
  const int nh   = wave >> 2;    // 32-col group within 64
  const int m0   = blockIdx.x * 128;
  const int n0   = blockIdx.y * 64;

  const unsigned sA0 = (unsigned)(size_t)&sA[0][0];   // LDS byte offset (low 32 bits of flat)
  const unsigned sB0 = (unsigned)(size_t)&sB[0][0];
  const bf16* agb = A  + (size_t)m0 * 1024;
  const bf16* bgb = Bt + (size_t)n0 * 1024;

  // issue one K-stage (64 wide) of async global->LDS copies: 6 x b128 per thread
  auto issue = [&](int k0, int buf) {
#pragma unroll
    for (int i = 0; i < 4; ++i) {                      // A: 128x64 bf16 = 16 KB
      const int seg = t + i * 256;
      const int row = seg >> 3;
      const int kk  = (seg & 7) * 8;
      const unsigned la = sA0 + (unsigned)((buf * 128 * 80 + row * 80 + kk) * 2);
      const unsigned long long ga =
          (unsigned long long)(size_t)(agb + (size_t)row * 1024 + k0 + kk);
      asm volatile("global_load_async_to_lds_b128 %0, %1, off" :: "v"(la), "v"(ga) : "memory");
    }
#pragma unroll
    for (int i = 0; i < 2; ++i) {                      // B: 64x64 bf16 = 8 KB
      const int seg = t + i * 256;
      const int row = seg >> 3;
      const int kk  = (seg & 7) * 8;
      const unsigned la = sB0 + (unsigned)((buf * 64 * 80 + row * 80 + kk) * 2);
      const unsigned long long ga =
          (unsigned long long)(size_t)(bgb + (size_t)row * 1024 + k0 + kk);
      asm volatile("global_load_async_to_lds_b128 %0, %1, off" :: "v"(la), "v"(ga) : "memory");
    }
  };

  v8f acc[2][2];   // [mt2][nt]
#pragma unroll
  for (int i = 0; i < 2; ++i)
#pragma unroll
    for (int j = 0; j < 2; ++j) acc[i][j] = (v8f){};

  issue(0, 0);
  for (int it = 0; it < 16; ++it) {
    const int buf = it & 1;
    if (it + 1 < 16) {
      issue((it + 1) * 64, buf ^ 1);
      // async loads complete in order per wave: <=6 outstanding drains the old stage
      asm volatile("s_wait_asynccnt 0x6" ::: "memory");
    } else {
      asm volatile("s_wait_asynccnt 0x0" ::: "memory");
    }
    __syncthreads();

    const bf16* a = &sA[buf][0];
    const bf16* b = &sB[buf][0];
#pragma unroll
    for (int ks = 0; ks < 2; ++ks) {
      // B fragments: column n, K = ks*32 + hf*16 + {0..15} contiguous
      v16bf bf0 = *(const v16bf*)&b[(nh * 32 + l15) * 80 + ks * 32 + hf * 16];
      v16bf bf1 = *(const v16bf*)&b[(nh * 32 + 16 + l15) * 80 + ks * 32 + hf * 16];
#pragma unroll
      for (int mt2 = 0; mt2 < 2; ++mt2) {
        const int arow = mq * 32 + mt2 * 16 + l15;
        v16bf af;   // A fragment: K = ks*32 + hf*16 + {0..15} contiguous
        ((v8bf*)&af)[0] = *(const v8bf*)&a[arow * 80 + ks * 32 + hf * 16];
        ((v8bf*)&af)[1] = *(const v8bf*)&a[arow * 80 + ks * 32 + hf * 16 + 8];
        acc[mt2][0] = wmma_bf16(af, bf0, acc[mt2][0]);
        acc[mt2][1] = wmma_bf16(af, bf1, acc[mt2][1]);
      }
    }
    __syncthreads();
  }

  const int d0 = nh * 32 + l15;
  const int d1 = d0 + 16;
  const float b0v = bias[n0 + d0];
  const float b1v = bias[n0 + d1];

  if constexpr (MODE == 3) {
#pragma unroll
    for (int mt2 = 0; mt2 < 2; ++mt2)
#pragma unroll
      for (int r = 0; r < 8; ++r) {
        const int ml = mq * 32 + mt2 * 16 + 8 * hf + r;
        const size_t m = (size_t)m0 + ml;
        of[m * (size_t)N + n0 + d0] = acc[mt2][0][r] + b0v;
        of[m * (size_t)N + n0 + d1] = acc[mt2][1][r] + b1v;
      }
  } else if constexpr (MODE == 2) {
    const int hd = blockIdx.y;
#pragma unroll
    for (int mt2 = 0; mt2 < 2; ++mt2)
#pragma unroll
      for (int r = 0; r < 8; ++r) {
        const int ml = mq * 32 + mt2 * 16 + 8 * hf + r;
        const size_t m = (size_t)m0 + ml;
        const int bb = (int)(m >> 12);
        const int s  = (int)(m & 4095);
        obf[(((size_t)bb * NKV + hd) * 64 + d0) * SEQ + s] = (bf16)(acc[mt2][0][r] + b0v);
        obf[(((size_t)bb * NKV + hd) * 64 + d1) * SEQ + s] = (bf16)(acc[mt2][1][r] + b1v);
      }
  } else {
#pragma unroll
    for (int mt2 = 0; mt2 < 2; ++mt2)
#pragma unroll
      for (int r = 0; r < 8; ++r) {
        const int ml = mq * 32 + mt2 * 16 + 8 * hf + r;
        sC[ml * 68 + d0] = acc[mt2][0][r] + b0v;
        sC[ml * 68 + d1] = acc[mt2][1][r] + b1v;
      }
    __syncthreads();

    const int row  = t >> 1;     // 0..127
    const int part = t & 1;      // 32 cols each
    const float* rp = &sC[row * 68];
    float ss = 0.f;
#pragma unroll
    for (int i = 0; i < 32; ++i) { float v = rp[part * 32 + i]; ss += v * v; }
    ss += __shfl_xor(ss, 1, 2);
    const float rs = rsqrtf(ss * (1.f / 64.f) + 1e-6f);

    const size_t m = (size_t)m0 + row;
    const int bb = (int)(m >> 12);
    const int s  = (int)(m & 4095);
    const int hd = blockIdx.y;
    constexpr int HEADS = (MODE == 0) ? NH : NKV;
    bf16* dst = obf + (((size_t)bb * HEADS + hd) * SEQ + s) * 64;
    const float* cr = cosT + (size_t)s * 64;
    const float* sr = sinT + (size_t)s * 64;
#pragma unroll
    for (int i = 0; i < 32; ++i) {
      const int d  = part * 32 + i;
      const int d2 = (d + 32) & 63;
      const float v   = rp[d]  * rs * nw[d];
      const float v2  = rp[d2] * rs * nw[d2];
      const float rot = (d < 32) ? -v2 : v2;
      dst[d] = (bf16)(v * cr[d] + rot * sr[d]);
    }
  }
}

// ---------------- flash-style chunked sliding-window attention ----------------
// grid: B*NH*NCH blocks, 256 threads (8 waves); wave owns 32 queries.
__global__ __launch_bounds__(256, 1)
void attn_kernel(const bf16* __restrict__ q, const bf16* __restrict__ k,
                 const bf16* __restrict__ vT, bf16* __restrict__ out) {
  __shared__ __align__(32) bf16 sp[8][2][16][32];   // per-wave P tiles (bf16)

  const int bid = blockIdx.x;
  const int c  = bid & 15;
  const int h  = (bid >> 4) & 15;
  const int b  = bid >> 8;
  const int hk = h >> 2;

  const bf16* qp = q  + ((size_t)b * NH  + h ) * SEQ * 64;
  const bf16* kp = k  + ((size_t)b * NKV + hk) * SEQ * 64;
  const bf16* vp = vT + ((size_t)b * NKV + hk) * 64 * SEQ;

  const int wave = threadIdx.x >> 5;
  const int lane = threadIdx.x & 31;
  const int hf   = lane >> 4;
  const int l15  = lane & 15;
  const int q0   = c * CHUNK + wave * 32;

  // cache Q fragments (A layout) for both m-tiles and both K-steps over DH
  v16bf qf[2][2];
#pragma unroll
  for (int mt = 0; mt < 2; ++mt) {
    const int row = q0 + mt * 16 + l15;
#pragma unroll
    for (int ks = 0; ks < 2; ++ks) {
      const size_t base = (size_t)row * 64 + ks * 32 + hf * 16;
      ((v8bf*)&qf[mt][ks])[0] = *(const v8bf*)&qp[base];
      ((v8bf*)&qf[mt][ks])[1] = *(const v8bf*)&qp[base + 8];
    }
  }

  v8f oacc[2][4];
#pragma unroll
  for (int mt = 0; mt < 2; ++mt)
#pragma unroll
    for (int nd = 0; nd < 4; ++nd) oacc[mt][nd] = (v8f){};
  float mrun[2][8], lrun[2][8];
#pragma unroll
  for (int mt = 0; mt < 2; ++mt)
#pragma unroll
    for (int r = 0; r < 8; ++r) { mrun[mt][r] = -3e38f; lrun[mt][r] = 0.f; }

  const float scale = 0.125f;   // 1/sqrt(64)
  const int kgbase = (c - 1) * CHUNK;
  const int wlo = q0 - HALFW;
  const int whi = q0 + 31 + HALFW;

  for (int jt = 0; jt < 24; ++jt) {
    const int kg0 = kgbase + jt * 32;
    if (kg0 + 31 < wlo || kg0 > whi) continue;       // outside window for this wave
    if (kg0 + 31 < 0 || kg0 >= SEQ) continue;        // outside sequence

    // prefetch next key tile (speculative)
    if (kg0 + 32 + 31 <= whi && kg0 + 32 < SEQ)
      __builtin_prefetch(&kp[(size_t)(kg0 + 32) * 64 + lane * 32], 0, 1);

    // K fragments (B layout): n = key, K(=dh) = ks*32 + hf*16 + 0..15 contiguous
    v16bf kf[2][2];
#pragma unroll
    for (int nt = 0; nt < 2; ++nt) {
      int key = kg0 + nt * 16 + l15;
      key = key < 0 ? 0 : (key > SEQ - 1 ? SEQ - 1 : key);
#pragma unroll
      for (int ks = 0; ks < 2; ++ks)
        kf[nt][ks] = *(const v16bf*)&kp[(size_t)key * 64 + ks * 32 + hf * 16];
    }

#pragma unroll
    for (int mt = 0; mt < 2; ++mt) {
      v8f sc0 = {}, sc1 = {};
      sc0 = wmma_bf16(qf[mt][0], kf[0][0], sc0);
      sc0 = wmma_bf16(qf[mt][1], kf[0][1], sc0);
      sc1 = wmma_bf16(qf[mt][0], kf[1][0], sc1);
      sc1 = wmma_bf16(qf[mt][1], kf[1][1], sc1);

#pragma unroll
      for (int r = 0; r < 8; ++r) {
        const int row = q0 + mt * 16 + 8 * hf + r;
        const int ka = kg0 + l15;
        const int kb = ka + 16;
        const bool oa = (ka >= row - HALFW) && (ka <= row + HALFW) && (ka >= 0) && (ka < SEQ);
        const bool ob = (kb >= row - HALFW) && (kb <= row + HALFW) && (kb >= 0) && (kb < SEQ);
        float a  = oa ? sc0[r] * scale : -3e38f;
        float bb = ob ? sc1[r] * scale : -3e38f;
        float tmax = fmaxf(a, bb);
        tmax = fmaxf(tmax, __shfl_xor(tmax, 1, 16));
        tmax = fmaxf(tmax, __shfl_xor(tmax, 2, 16));
        tmax = fmaxf(tmax, __shfl_xor(tmax, 4, 16));
        tmax = fmaxf(tmax, __shfl_xor(tmax, 8, 16));
        const float mo = mrun[mt][r];
        const float mn = fmaxf(mo, tmax);
        float corr, pa, pb;
        if (mn < -1e30f) { corr = 1.f; pa = 0.f; pb = 0.f; }
        else {
          corr = (mo < -1e30f) ? 0.f : __expf(mo - mn);
          pa = oa ? __expf(a - mn) : 0.f;
          pb = ob ? __expf(bb - mn) : 0.f;
        }
        float rsum = pa + pb;
        rsum += __shfl_xor(rsum, 1, 16);
        rsum += __shfl_xor(rsum, 2, 16);
        rsum += __shfl_xor(rsum, 4, 16);
        rsum += __shfl_xor(rsum, 8, 16);
        lrun[mt][r] = lrun[mt][r] * corr + rsum;
        mrun[mt][r] = mn;
#pragma unroll
        for (int nd = 0; nd < 4; ++nd) oacc[mt][nd][r] *= corr;
        const int lm = 8 * hf + r;
        sp[wave][mt][lm][l15]      = (bf16)pa;
        sp[wave][mt][lm][l15 + 16] = (bf16)pb;
      }
    }

    // same-wave LDS store -> load turnaround
    asm volatile("s_wait_dscnt 0" ::: "memory");

    // V^T fragments (B layout): n = dh, K = key offset hf*16 + 0..15 contiguous
    int key0 = kg0 + hf * 16;
    key0 = key0 < 0 ? 0 : (key0 > SEQ - 16 ? SEQ - 16 : key0);
    v16bf vf[4];
#pragma unroll
    for (int nd = 0; nd < 4; ++nd) {
      const int dh = nd * 16 + l15;
      vf[nd] = *(const v16bf*)&vp[(size_t)dh * SEQ + key0];
    }

#pragma unroll
    for (int mt = 0; mt < 2; ++mt) {
      const int kofs = hf * 16;
      v16bf pf;
      ((v8bf*)&pf)[0] = *(const v8bf*)&sp[wave][mt][l15][kofs];
      ((v8bf*)&pf)[1] = *(const v8bf*)&sp[wave][mt][l15][kofs + 8];
#pragma unroll
      for (int nd = 0; nd < 4; ++nd)
        oacc[mt][nd] = wmma_bf16(pf, vf[nd], oacc[mt][nd]);
    }
  }

  // epilogue: normalize and scatter bf16 to [B*S, H*64]
#pragma unroll
  for (int mt = 0; mt < 2; ++mt) {
#pragma unroll
    for (int r = 0; r < 8; ++r) {
      const float l = lrun[mt][r];
      const float inv = l > 0.f ? 1.f / l : 0.f;
      const int row = q0 + mt * 16 + 8 * hf + r;
      bf16* dst = out + ((size_t)b * SEQ + row) * (NH * 64) + h * 64;
#pragma unroll
      for (int nd = 0; nd < 4; ++nd)
        dst[nd * 16 + l15] = (bf16)(oacc[mt][nd][r] * inv);
    }
  }
}

// ---------------- host launch ----------------
extern "C" void kernel_launch(void* const* d_in, const int* in_sizes, int n_in,
                              void* d_out, int out_size, void* d_ws, size_t ws_size,
                              hipStream_t stream) {
  const float* x    = (const float*)d_in[0];
  const float* Wq   = (const float*)d_in[1];
  const float* bq   = (const float*)d_in[2];
  const float* Wk   = (const float*)d_in[3];
  const float* bk   = (const float*)d_in[4];
  const float* Wv   = (const float*)d_in[5];
  const float* bv   = (const float*)d_in[6];
  const float* Wo   = (const float*)d_in[7];
  const float* bo   = (const float*)d_in[8];
  const float* qnw  = (const float*)d_in[9];
  const float* knw  = (const float*)d_in[10];
  const float* cosT = (const float*)d_in[11];
  const float* sinT = (const float*)d_in[12];

  const size_t M = (size_t)BQ * SEQ;           // 16384
  char* ws = (char*)d_ws;
  size_t off = 0;
  auto alloc = [&](size_t bytes) { size_t o = off; off = (off + bytes + 255) & ~(size_t)255; return o; };

  const size_t xb_o  = alloc(M * DMODEL * 2);                 // x bf16; reused as attn-out bf16
  const size_t wqT_o = alloc((size_t)DMODEL * 1024 * 2);
  const size_t wkT_o = alloc((size_t)DMODEL * 256 * 2);
  const size_t wvT_o = alloc((size_t)DMODEL * 256 * 2);
  const size_t woT_o = alloc((size_t)DMODEL * 1024 * 2);
  const size_t qb_o  = alloc((size_t)BQ * NH  * SEQ * 64 * 2);
  const size_t kb_o  = alloc((size_t)BQ * NKV * SEQ * 64 * 2);
  const size_t vT_o  = alloc((size_t)BQ * NKV * 64 * SEQ * 2);
  if (off > ws_size) return;  // workspace too small (deterministic no-op)

  bf16* xb  = (bf16*)(ws + xb_o);
  bf16* wqT = (bf16*)(ws + wqT_o);
  bf16* wkT = (bf16*)(ws + wkT_o);
  bf16* wvT = (bf16*)(ws + wvT_o);
  bf16* woT = (bf16*)(ws + woT_o);
  bf16* qb  = (bf16*)(ws + qb_o);
  bf16* kb  = (bf16*)(ws + kb_o);
  bf16* vTb = (bf16*)(ws + vT_o);
  bf16* aob = xb;                      // reuse x-bf16 region for attention output
  float* outf = (float*)d_out;

  // 1) conversions
  f2bf_kernel<<<8192, 256, 0, stream>>>(x, xb, M * DMODEL);
  transpose_f2bf_kernel<<<4096, 256, 0, stream>>>(Wq, wqT, DMODEL, 1024);
  transpose_f2bf_kernel<<<1024, 256, 0, stream>>>(Wk, wkT, DMODEL, 256);
  transpose_f2bf_kernel<<<1024, 256, 0, stream>>>(Wv, wvT, DMODEL, 256);
  transpose_f2bf_kernel<<<4096, 256, 0, stream>>>(Wo, woT, DMODEL, 1024);

  // 2) projections (+norm/rope fused)
  gemm128_kernel<0><<<dim3(M / 128, 1024 / 64), 256, 0, stream>>>(xb, wqT, bq, qnw, cosT, sinT, qb, nullptr, 1024);
  gemm128_kernel<1><<<dim3(M / 128, 256 / 64),  256, 0, stream>>>(xb, wkT, bk, knw, cosT, sinT, kb, nullptr, 256);
  gemm128_kernel<2><<<dim3(M / 128, 256 / 64),  256, 0, stream>>>(xb, wvT, bv, nullptr, nullptr, nullptr, vTb, nullptr, 256);

  // 3) attention (writes aob, which aliases xb -- xb no longer needed)
  attn_kernel<<<BQ * NH * NCH, 256, 0, stream>>>(qb, kb, vTb, aob);

  // 4) output projection -> fp32 d_out
  gemm128_kernel<3><<<dim3(M / 128, 1024 / 64), 256, 0, stream>>>(aob, woT, bo, nullptr, nullptr, nullptr, nullptr, outf, 1024);
}
